// PolicyNet_74809740361771
// MI455X (gfx1250) — matrix operand order repaired and checked by reading, not compiled, not run on hardware
//
#include <hip/hip_runtime.h>
#include <hip/hip_bf16.h>

typedef __bf16 v16bf __attribute__((ext_vector_type(16)));
typedef float  v8f   __attribute__((ext_vector_type(8)));
typedef unsigned short u16x8  __attribute__((ext_vector_type(8)));
typedef unsigned short u16x16 __attribute__((ext_vector_type(16)));

#define DIM 128
#define LQ 16
#define WAVES 2            // waves per block (wave32)
#define MT 2               // 16-row M-tiles per wave -> 32 actions per wave
#define ROWS (MT * 16)     // rows per wave
#define BLK_T (WAVES * 32) // threads per block
#define BLK_A (WAVES * ROWS)

__device__ __forceinline__ unsigned short f2bf(float f) {
    unsigned int u = __builtin_bit_cast(unsigned int, f);
    u += 0x7fffu + ((u >> 16) & 1u);          // round-to-nearest-even
    return (unsigned short)(u >> 16);
}

__device__ __forceinline__ v16bf as_bf16(u16x16 u) {
    union { u16x16 u; v16bf b; } c; c.u = u; return c.b;
}

// fast tanh: 1 - 2/(exp2(2*log2e*x)+1)  -> v_exp_f32 + v_rcp_f32 (trans ops, co-exec with WMMA)
__device__ __forceinline__ float fast_tanh(float x) {
    float e = __builtin_amdgcn_exp2f(x * 2.885390081777927f);
    return 1.0f - 2.0f * __builtin_amdgcn_rcpf(e + 1.0f);
}

// A-matrix fragment (16x32 bf16) from row-major bf16 rows in LDS.
// lane<16: row=lane, K = [0..15]; lane>=16: row=lane-16, K = [16..31]
__device__ __forceinline__ v16bf load_a(const unsigned short* buf, int l16, int half, int ks) {
    const u16x8* p = (const u16x8*)(buf + l16 * DIM + ks * 32 + half * 16);
    u16x8 a0 = p[0];
    u16x8 a1 = p[1];   // +8 ushorts
    return as_bf16(__builtin_shufflevector(a0, a1, 0,1,2,3,4,5,6,7,8,9,10,11,12,13,14,15));
}

// B-matrix fragment (32x16 bf16) from column-major (N-major) bf16 weights in global.
// lane n = lane%16; K = (half?16:0)+ks*32 .. +15 contiguous.
__device__ __forceinline__ v16bf load_b(const unsigned short* w, int l16, int half, int nt, int ks) {
    const u16x8* p = (const u16x8*)(w + (nt * 16 + l16) * DIM + ks * 32 + half * 16);
    u16x8 b0 = p[0];
    u16x8 b1 = p[1];
    return as_bf16(__builtin_shufflevector(b0, b1, 0,1,2,3,4,5,6,7,8,9,10,11,12,13,14,15));
}

#define WMMA_BF16(A, B, C) \
    __builtin_amdgcn_wmma_f32_16x16x32_bf16(false, (A), false, (B), (short)0, (C), false, false)

// ---------------- precompute: bf16-transpose weights, fold biases -------------
__global__ __launch_bounds__(256) void policy_precompute(
    const float* __restrict__ question, const float* __restrict__ history,
    const float* __restrict__ attn_W, const float* __restrict__ attn_b,
    const float* __restrict__ p_W1, const float* __restrict__ p_b1,
    const float* __restrict__ p_W2,
    unsigned short* __restrict__ WtA, unsigned short* __restrict__ Wt1,
    unsigned short* __restrict__ Wt2,
    float* __restrict__ v_out, float* __restrict__ c_out)
{
    int tid = threadIdx.x;
    // transpose to [N][K] bf16
    for (int idx = tid; idx < DIM * DIM; idx += 256) {
        int n = idx >> 7, k = idx & 127;
        WtA[n * DIM + k] = f2bf(attn_W[k * DIM + n]);            // top half of attn_W
        Wt1[n * DIM + k] = f2bf(p_W1[(DIM + k) * DIM + n]);      // bottom half of p_W1
        Wt2[n * DIM + k] = f2bf(p_W2[k * DIM + n]);
    }
    // v[l][n] = question[l] . attn_W[128:,n] + attn_b[n]
    for (int o = tid; o < LQ * DIM; o += 256) {
        int l = o >> 7, n = o & 127;
        float s = attn_b[n];
        for (int k = 0; k < DIM; ++k) s += question[l * DIM + k] * attn_W[(DIM + k) * DIM + n];
        v_out[o] = s;
    }
    // c[n] = history . p_W1[:128,n] + p_b1[n]
    for (int n = tid; n < DIM; n += 256) {
        float s = p_b1[n];
        for (int k = 0; k < DIM; ++k) s += history[k] * p_W1[k * DIM + n];
        c_out[n] = s;
    }
}

// ---------------- main: gather + 3 WMMA GEMMs + tanh/relu epilogues ----------
// Each wave owns MT=2 16-row tiles; each B fragment feeds both tiles' WMMAs
// (halves L2 weight traffic, interleaves two independent accumulator chains).
__global__ __launch_bounds__(BLK_T) void policy_main(
    const int* __restrict__ actions, const float* __restrict__ rel_table,
    const unsigned short* __restrict__ WtA, const unsigned short* __restrict__ Wt1,
    const unsigned short* __restrict__ Wt2,
    const float* __restrict__ v_g, const float* __restrict__ c_g,
    const float* __restrict__ b2_g, float* __restrict__ sem)
{
    __shared__ __align__(16) unsigned short Abuf[WAVES][ROWS * DIM]; // rel bf16, reused as H bf16
    __shared__ __align__(16) unsigned short Qbuf[WAVES][ROWS * DIM]; // q_sum bf16
    __shared__ float vls[LQ * DIM];
    __shared__ float cls[DIM];
    __shared__ float b2ls[DIM];
    __shared__ int   acts[WAVES][ROWS];

    const int tid  = threadIdx.x;
    const int w    = tid >> 5;
    const int lane = tid & 31;
    const int l16  = lane & 15;
    const int half = lane >> 4;
    const int abase = blockIdx.x * BLK_A + w * ROWS;

    for (int i = tid; i < LQ * DIM; i += BLK_T) vls[i] = v_g[i];
    for (int i = tid; i < DIM; i += BLK_T) { cls[i] = c_g[i]; b2ls[i] = b2_g[i]; }
    acts[w][lane] = actions[abase + lane];       // 32 rows per wave

    // gather ROWS rel rows per wave -> bf16 A-buffer (table is L2-resident)
    for (int r = 0; r < ROWS; ++r) {
        int row = actions[abase + r];
        const float4* src = (const float4*)(rel_table + (size_t)row * DIM);
        float4 d = src[lane];
        unsigned short* dst = &Abuf[w][r * DIM + lane * 4];
        dst[0] = f2bf(d.x); dst[1] = f2bf(d.y); dst[2] = f2bf(d.z); dst[3] = f2bf(d.w);
    }
    __syncthreads();

    float dotacc[MT][8];
#pragma unroll
    for (int t = 0; t < MT; ++t)
#pragma unroll
        for (int i = 0; i < 8; ++i) dotacc[t][i] = 0.0f;

    // ---- GEMM 1: u = rel @ W_top ; q_sum = sum_l tanh(u + v[l]) ----
    for (int nt = 0; nt < 8; ++nt) {
        v8f acc[MT];
#pragma unroll
        for (int t = 0; t < MT; ++t) acc[t] = v8f{};
#pragma unroll
        for (int ks = 0; ks < 4; ++ks) {
            v16bf b = load_b(WtA, l16, half, nt, ks);
#pragma unroll
            for (int t = 0; t < MT; ++t)
                acc[t] = WMMA_BF16(load_a(&Abuf[w][t * 16 * DIM], l16, half, ks), b, acc[t]);
        }
        const int n = nt * 16 + l16;
        float qs[MT][8];
#pragma unroll
        for (int t = 0; t < MT; ++t)
#pragma unroll
            for (int i = 0; i < 8; ++i) qs[t][i] = 0.0f;
        for (int l = 0; l < LQ; ++l) {
            float vv = vls[l * DIM + n];           // shared by both tiles
#pragma unroll
            for (int t = 0; t < MT; ++t)
#pragma unroll
                for (int i = 0; i < 8; ++i) qs[t][i] += fast_tanh(acc[t][i] + vv);
        }
#pragma unroll
        for (int t = 0; t < MT; ++t)
#pragma unroll
            for (int i = 0; i < 8; ++i)            // C-layout: m = i + 8*half
                Qbuf[w][(t * 16 + i + 8 * half) * DIM + n] = f2bf(qs[t][i]);
    }

    // ---- GEMM 2: h = relu(q_sum @ W1_bot + c) ----
    for (int nt = 0; nt < 8; ++nt) {
        v8f acc[MT];
#pragma unroll
        for (int t = 0; t < MT; ++t) acc[t] = v8f{};
#pragma unroll
        for (int ks = 0; ks < 4; ++ks) {
            v16bf b = load_b(Wt1, l16, half, nt, ks);
#pragma unroll
            for (int t = 0; t < MT; ++t)
                acc[t] = WMMA_BF16(load_a(&Qbuf[w][t * 16 * DIM], l16, half, ks), b, acc[t]);
        }
        const int n = nt * 16 + l16;
        float cv = cls[n];
#pragma unroll
        for (int t = 0; t < MT; ++t)
#pragma unroll
            for (int i = 0; i < 8; ++i) {
                float h = acc[t][i] + cv;
                Abuf[w][(t * 16 + i + 8 * half) * DIM + n] = f2bf(h > 0.0f ? h : 0.0f);
            }
    }

    // ---- GEMM 3: qwh = h @ W2 + b2 ; sem[m] = sum_n qwh[m,n]*rel[m,n] ----
    for (int nt = 0; nt < 8; ++nt) {
        v8f acc[MT];
#pragma unroll
        for (int t = 0; t < MT; ++t) acc[t] = v8f{};
#pragma unroll
        for (int ks = 0; ks < 4; ++ks) {
            v16bf b = load_b(Wt2, l16, half, nt, ks);
#pragma unroll
            for (int t = 0; t < MT; ++t)
                acc[t] = WMMA_BF16(load_a(&Abuf[w][t * 16 * DIM], l16, half, ks), b, acc[t]);
        }
        const int n = nt * 16 + l16;
        float bb = b2ls[n];
#pragma unroll
        for (int t = 0; t < MT; ++t)
#pragma unroll
            for (int i = 0; i < 8; ++i) {
                int m = t * 16 + i + 8 * half;
                float relv = rel_table[(size_t)acts[w][m] * DIM + n];   // L2 hit (just gathered)
                dotacc[t][i] += (acc[t][i] + bb) * relv;
            }
    }

    // reduce each row's partial dot across its 16-lane group
#pragma unroll
    for (int t = 0; t < MT; ++t)
#pragma unroll
        for (int i = 0; i < 8; ++i) {
            float s = dotacc[t][i];
            s += __shfl_xor(s, 1, 32);
            s += __shfl_xor(s, 2, 32);
            s += __shfl_xor(s, 4, 32);
            s += __shfl_xor(s, 8, 32);
            if (l16 == 0) sem[abase + t * 16 + i + 8 * half] = s; // lane0 -> m=i, lane16 -> m=8+i
        }
}

// ---------------- global softmax over A (in place on d_out) ------------------
__global__ __launch_bounds__(1024) void policy_softmax(float* __restrict__ buf, int n)
{
    __shared__ float red[1024];
    const int tid = threadIdx.x;
    float mx = -3.4e38f;
    for (int i = tid; i < n; i += 1024) mx = fmaxf(mx, buf[i]);
    red[tid] = mx; __syncthreads();
    for (int s = 512; s > 0; s >>= 1) {
        if (tid < s) red[tid] = fmaxf(red[tid], red[tid + s]);
        __syncthreads();
    }
    mx = red[0]; __syncthreads();
    float sum = 0.0f;
    for (int i = tid; i < n; i += 1024)
        sum += __builtin_amdgcn_exp2f((buf[i] - mx) * 1.4426950408889634f);
    red[tid] = sum; __syncthreads();
    for (int s = 512; s > 0; s >>= 1) {
        if (tid < s) red[tid] += red[tid + s];
        __syncthreads();
    }
    float inv = 1.0f / red[0];
    for (int i = tid; i < n; i += 1024)    // each element owned by exactly one thread: safe in place
        buf[i] = __builtin_amdgcn_exp2f((buf[i] - mx) * 1.4426950408889634f) * inv;
}

extern "C" void kernel_launch(void* const* d_in, const int* in_sizes, int n_in,
                              void* d_out, int out_size, void* d_ws, size_t ws_size,
                              hipStream_t stream) {
    const int*   actions   = (const int*)d_in[0];
    const float* question  = (const float*)d_in[1];
    const float* history   = (const float*)d_in[2];
    const float* rel_table = (const float*)d_in[3];
    const float* attn_W    = (const float*)d_in[4];
    const float* attn_b    = (const float*)d_in[5];
    const float* p_W1      = (const float*)d_in[6];
    const float* p_b1      = (const float*)d_in[7];
    const float* p_W2      = (const float*)d_in[8];
    const float* p_b2      = (const float*)d_in[9];
    float* out = (float*)d_out;

    const int A = in_sizes[0];            // 32768

    char* ws = (char*)d_ws;               // ~105 KB used
    unsigned short* WtA = (unsigned short*)(ws);
    unsigned short* Wt1 = (unsigned short*)(ws + 32768);
    unsigned short* Wt2 = (unsigned short*)(ws + 65536);
    float* v_g = (float*)(ws + 98304);    // 16*128 f32 (attn_b folded in)
    float* c_g = (float*)(ws + 106496);   // 128 f32

    policy_precompute<<<1, 256, 0, stream>>>(question, history, attn_W, attn_b,
                                             p_W1, p_b1, p_W2, WtA, Wt1, Wt2, v_g, c_g);
    policy_main<<<A / BLK_A, BLK_T, 0, stream>>>(
        actions, rel_table, WtA, Wt1, Wt2, v_g, c_g, p_b2, out /* sem scores */);
    policy_softmax<<<1, 1024, 0, stream>>>(out, A);
}
